// _ConvAttentionCoefficients_49168785605370
// MI455X (gfx1250) — matrix operand order repaired and checked by reading, not compiled, not run on hardware
//
#include <hip/hip_runtime.h>

typedef __attribute__((ext_vector_type(16))) _Float16 v16h;
typedef __attribute__((ext_vector_type(8)))  _Float16 v8h;
typedef __attribute__((ext_vector_type(8)))  float    v8f;

#define FEAT 128
#define KHALF 192            // K rows staged per LDS pass (2 x 192 = 384)
#define LDS_STRIDE 200       // halves per column row (400B -> perfect bank spread)
#define HALF_ELEMS (FEAT * LDS_STRIDE)   // 25600 halves = 51200 B per K-half
#define PAIRS_PER_BLOCK 256  // 8 waves x 32 pairs
#define THREADS 256

// ---------------------------------------------------------------------------
// One-shot prep: W1 (384x128 f32, row-major [k][col]) -> f16, transposed and
// bank-padded into d_ws as two K-halves: ws[half][col][0..199] (k' = k%192).
// ---------------------------------------------------------------------------
__global__ __launch_bounds__(THREADS) void prep_w1_kernel(
    const float* __restrict__ W1, _Float16* __restrict__ w1t)
{
    const int linear = blockIdx.x * THREADS + threadIdx.x;   // 0..49151
    if (linear < KHALF * 2 * FEAT) {
        const int k   = linear >> 7;          // 0..383  (coalesced read)
        const int col = linear & (FEAT - 1);
        const int half = k / KHALF;
        const int kk   = k - half * KHALF;    // 0..191
        w1t[half * HALF_ELEMS + col * LDS_STRIDE + kk] = (_Float16)W1[linear];
    }
}

// Build one 16x32 f16 A fragment per the ISA layout (lane = m + 16*s):
// VGPR j holds K = (j>>2)*16 + s*8 + (j&3)*2 (+1). kin = (ks&3)*32 + s*8.
__device__ __forceinline__ v16h load_a_frag(const float* __restrict__ src, int kin)
{
    v16h a;
    #pragma unroll
    for (int g = 0; g < 2; ++g) {
        const float4 d0 = *(const float4*)(src + kin + g * 16);
        const float4 d1 = *(const float4*)(src + kin + g * 16 + 4);
        a[8*g + 0] = (_Float16)d0.x;  a[8*g + 1] = (_Float16)d0.y;
        a[8*g + 2] = (_Float16)d0.z;  a[8*g + 3] = (_Float16)d0.w;
        a[8*g + 4] = (_Float16)d1.x;  a[8*g + 5] = (_Float16)d1.y;
        a[8*g + 6] = (_Float16)d1.z;  a[8*g + 7] = (_Float16)d1.w;
    }
    return a;
}

// Epilogue for one 16-pair tile: silu(acc + b1) . W2, reduce 16 lanes, store.
#define TILE_EPILOGUE(ACC, ROW0)                                              \
    {                                                                         \
        _Pragma("unroll")                                                     \
        for (int e = 0; e < 8; ++e) {                                         \
            float s = 0.f;                                                    \
            _Pragma("unroll")                                                 \
            for (int nt = 0; nt < 8; ++nt) {                                  \
                const float v = ACC[nt][e] + b1v[nt];                         \
                s += (v / (1.0f + __expf(-v))) * w2v[nt];                     \
            }                                                                 \
            s += __shfl_xor(s, 1);                                            \
            s += __shfl_xor(s, 2);                                            \
            s += __shfl_xor(s, 4);                                            \
            s += __shfl_xor(s, 8);                                            \
            const int rowOut = (ROW0) + 8 * t + e;                            \
            if (n == e && rowOut < nPairs) out[rowOut] = s + bias2;           \
        }                                                                     \
    }

__global__ __launch_bounds__(THREADS) void edge_mlp_kernel(
    const float* __restrict__ x, const float* __restrict__ w_ij,
    const _Float16* __restrict__ w1t, const float* __restrict__ b1,
    const float* __restrict__ W2, const float* __restrict__ b2,
    const int* __restrict__ idx_i, const int* __restrict__ idx_j,
    float* __restrict__ out, int nPairs)
{
    __shared__ _Float16 ldsB[HALF_ELEMS];   // 51,200 bytes

    const int tid  = threadIdx.x;
    const int wave = tid >> 5;
    const int lane = tid & 31;
    const int n    = lane & 15;   // M row (A) / N col (B,C,D) within tile
    const int t    = lane >> 4;   // half-wave selector (s for A, t for B/C/D)

    const int waveBase = blockIdx.x * PAIRS_PER_BLOCK + wave * 32;
    int r0 = waveBase + n;                          // tile-0 pair row
    int r1 = waveBase + 16 + n;                     // tile-1 pair row
    if (r0 >= nPairs) r0 = nPairs - 1;
    if (r1 >= nPairs) r1 = nPairs - 1;

    // Per-lane source row pointers: y = [ x[idx_i] | x[idx_j] | w_ij ]
    const float* q0 = x + (long)idx_i[r0] * FEAT;
    const float* k0 = x + (long)idx_j[r0] * FEAT;
    const float* w0 = w_ij + (long)r0 * FEAT;
    const float* q1 = x + (long)idx_i[r1] * FEAT;
    const float* k1 = x + (long)idx_j[r1] * FEAT;
    const float* w1p = w_ij + (long)r1 * FEAT;

    // Pull the HBM-resident w_ij rows early (x rows are L2-resident: 25.6MB).
    __builtin_prefetch(w0, 0, 1);
    __builtin_prefetch(w0 + 64, 0, 1);
    __builtin_prefetch(w1p, 0, 1);
    __builtin_prefetch(w1p + 64, 0, 1);

    v8f acc0[8] = {};   // tile 0: 16(M) x 128(N) f32 accumulators
    v8f acc1[8] = {};   // tile 1

    #pragma unroll
    for (int half = 0; half < 2; ++half) {
        __syncthreads();
        // ---- stage pre-transposed f16 W1 half: contiguous 8B copies ----
        {
            const uint2* src = (const uint2*)(w1t + half * HALF_ELEMS);
            uint2* dst = (uint2*)ldsB;
            #pragma unroll 5
            for (int i = 0; i < HALF_ELEMS / 4 / THREADS; ++i)   // 25 iters
                dst[i * THREADS + tid] = src[i * THREADS + tid];
        }
        __syncthreads();

        #pragma unroll
        for (int ks6 = 0; ks6 < 6; ++ks6) {
            const int ksg = half * 6 + ks6;                 // global k-step 0..11
            const float* sA0 = (ksg < 4) ? q0 : ((ksg < 8) ? k0 : w0);
            const float* sA1 = (ksg < 4) ? q1 : ((ksg < 8) ? k1 : w1p);
            const int kin = (ksg & 3) * 32 + t * 8;

            const v16h a0 = load_a_frag(sA0, kin);
            const v16h a1 = load_a_frag(sA1, kin);

            #pragma unroll
            for (int nt = 0; nt < 8; ++nt) {
                // B fragment: lane n+16t, element s holds K = t*8 + s (s<8), 16 + t*8 + (s-8) (s>=8)
                const _Float16* bp = &ldsB[(nt * 16 + n) * LDS_STRIDE
                                           + ks6 * 32 + t * 8];
                const v8h blo = *(const v8h*)(bp);          // 16B aligned
                const v8h bhi = *(const v8h*)(bp + 16);
                v16h b;
                #pragma unroll
                for (int q = 0; q < 8; ++q) { b[q] = blo[q]; b[q + 8] = bhi[q]; }

                acc0[nt] = __builtin_amdgcn_wmma_f32_16x16x32_f16(
                    false, a0, false, b, (short)0, acc0[nt], false, false);
                acc1[nt] = __builtin_amdgcn_wmma_f32_16x16x32_f16(
                    false, a1, false, b, (short)0, acc1[nt], false, false);
            }
        }
    }

    // ---- epilogue: out[m] = b2 + sum_c silu(acc[m,c] + b1[c]) * W2[c] ----
    float b1v[8], w2v[8];
    #pragma unroll
    for (int nt = 0; nt < 8; ++nt) {
        const int c = nt * 16 + n;
        b1v[nt] = b1[c];
        w2v[nt] = W2[c];
    }
    const float bias2 = b2[0];

    TILE_EPILOGUE(acc0, waveBase)
    TILE_EPILOGUE(acc1, waveBase + 16)
}

extern "C" void kernel_launch(void* const* d_in, const int* in_sizes, int n_in,
                              void* d_out, int out_size, void* d_ws, size_t ws_size,
                              hipStream_t stream) {
    const float* x    = (const float*)d_in[0];
    const float* w_ij = (const float*)d_in[1];
    const float* W1   = (const float*)d_in[2];
    const float* b1   = (const float*)d_in[3];
    const float* W2   = (const float*)d_in[4];
    const float* b2   = (const float*)d_in[5];
    const int*   idxi = (const int*)d_in[6];
    const int*   idxj = (const int*)d_in[7];
    float* out = (float*)d_out;
    _Float16* w1t = (_Float16*)d_ws;          // needs 102,400 B of scratch

    const int nPairs = in_sizes[6];   // 800000

    // One-shot W1 transpose/convert into scratch (same stream -> ordered).
    const int prepBlocks = (KHALF * 2 * FEAT + THREADS - 1) / THREADS;  // 192
    hipLaunchKernelGGL(prep_w1_kernel, dim3(prepBlocks), dim3(THREADS), 0, stream,
                       W1, w1t);

    const int blocks = (nPairs + PAIRS_PER_BLOCK - 1) / PAIRS_PER_BLOCK; // 3125
    hipLaunchKernelGGL(edge_mlp_kernel, dim3(blocks), dim3(THREADS), 0, stream,
                       x, w_ij, w1t, b1, W2, b2, idxi, idxj, out, nPairs);
}
